// GATLayer_58995670778248
// MI455X (gfx1250) — matrix operand order repaired and checked by reading, not compiled, not run on hardware
//
#include <hip/hip_runtime.h>
#include <hip/hip_bf16.h>

// ---------------------------------------------------------------------------
// GAT-style layer on MI455X (gfx1250, wave32, WMMA bf16 16x16x32).
// hsrc=src^T@x, htgt=tgt^T@x, y=relu([hsrc|htgt]@fw^T+fb), out=tgt@y
// HBM floor ~69us (src once + tgt twice @ 23.3TB/s); bf16 WMMA keeps math
// (~105 GFLOP) well under that, so the kernels are built to stream src/tgt
// (non-temporal) while x/fw/y fragments stay hot in the 192MB L2.
// ---------------------------------------------------------------------------

#define Nn 4096
#define Ee 32768
#define DI 128
#define DO 128
#define DK 256

typedef __attribute__((ext_vector_type(16))) __bf16 v16bf;
typedef __attribute__((ext_vector_type(8)))  float  v8f;
typedef __attribute__((ext_vector_type(4)))  float  f4;

// A-operand (16-bit 16x32) per-lane element->K mapping (interleaved):
//   K(i,hi) = (i&7) + hi*8 + ((i&8)<<1)
// B-operand (16-bit 32x16) per-lane element->K mapping (same interleaved order as A):
//   K(i,hi) = (i&7) + hi*8 + ((i&8)<<1) ; N = lane&15
__device__ __forceinline__ int akmap(int i, int hi) {
  return (i & 7) + hi * 8 + ((i & 8) << 1);
}

// ---------------------------------------------------------------------------
// Kernel 1: pack x [N,DI] and fw [DO,DK] into B-fragment bf16 layouts.
// xfrag : [N/32][DI/16][32 lanes][16]   fwfrag : [DK/32][DO/16][32][16]
// ---------------------------------------------------------------------------
__global__ __launch_bounds__(256) void prep_frags(
    const float* __restrict__ x, const float* __restrict__ fw,
    __bf16* __restrict__ xfrag, __bf16* __restrict__ fwfrag) {
  int id = blockIdx.x * 256 + threadIdx.x;
  const int NXF = (Nn / 32) * (DI / 16) * 32 * 16;   // 524288
  const int NWF = (DK / 32) * (DO / 16) * 32 * 16;   // 32768
  if (id < NXF) {
    int i = id & 15, lam = (id >> 4) & 31, dt = (id >> 9) & 7, nb = id >> 12;
    int n = nb * 32 + (i & 7) + ((lam >> 4) << 3) + ((i & 8) << 1);
    int d = dt * 16 + (lam & 15);
    xfrag[id] = (__bf16)x[n * DI + d];
  } else if (id < NXF + NWF) {
    int id2 = id - NXF;
    int i = id2 & 15, lam = (id2 >> 4) & 31, dt = (id2 >> 9) & 7, kb = id2 >> 12;
    int kk = kb * 32 + (i & 7) + ((lam >> 4) << 3) + ((i & 8) << 1);
    int d = dt * 16 + (lam & 15);
    fwfrag[id2] = (__bf16)fw[d * DK + kk];
  }
}

// ---------------------------------------------------------------------------
// Kernel 2 (fused): per 64-edge block:
//   waves 0-3: hsrc strips (16e x 128d), waves 4-7: htgt strips, K=N=4096
//   then h(64x256) via LDS -> y = relu(h@fw^T+fb) via WMMA ->
//   y written to global in B-fragment layout for the scatter GEMM.
// ---------------------------------------------------------------------------
__global__ __launch_bounds__(256) void gather_mlp(
    const float* __restrict__ src, const float* __restrict__ tgt,
    const float* __restrict__ fb, const __bf16* __restrict__ xfrag,
    const __bf16* __restrict__ fwfrag, __bf16* __restrict__ yfrag) {
  __shared__ __bf16 h_lds[64][DK];   // 32 KB
  __shared__ __bf16 y_lds[64][DO];   // 16 KB

  const int tid  = threadIdx.x;
  const int w    = tid >> 5;
  const int lane = tid & 31;
  const int m    = lane & 15;
  const int hi   = lane >> 4;
  const int half = w >> 2;           // 0 = src, 1 = tgt
  const int s    = w & 3;            // e-strip within block
  const int e0   = blockIdx.x * 64 + s * 16;
  const float* __restrict__ M = half ? tgt : src;

  v8f acc[8] = {};
  for (int nb = 0; nb < Nn / 32; ++nb) {
    const int n0 = nb * 32;
    // A tile (16e x 32n): A[m][k] = M[(n0+k)*E + e0+m]; coalesced along e.
    v16bf a;
#pragma unroll
    for (int i = 0; i < 16; ++i) {
      int k = akmap(i, hi);
      a[i] = (__bf16)__builtin_nontemporal_load(&M[(size_t)(n0 + k) * Ee + e0 + m]);
    }
#pragma unroll
    for (int t = 0; t < 8; ++t) {
      v16bf b = *(const v16bf*)(xfrag + (((size_t)nb * 8 + t) * 32 + lane) * 16);
      acc[t] = __builtin_amdgcn_wmma_f32_16x16x32_bf16(
          false, a, false, b, (short)0, acc[t], false, false);
    }
  }

  // Spill h strips (C layout: VGPR g -> row g + hi*8, col = m) to LDS as bf16.
#pragma unroll
  for (int t = 0; t < 8; ++t)
#pragma unroll
    for (int g = 0; g < 8; ++g)
      h_lds[s * 16 + g + hi * 8][half * 128 + t * 16 + m] = (__bf16)acc[t][g];
  __syncthreads();

  // MLP GEMM: wave w -> e-strip (w&3), d-half (w>>2): 4 dtiles, K=256.
  const int s2 = w & 3, hd = w >> 2;
  v8f yacc[4] = {};
#pragma unroll
  for (int kb = 0; kb < DK / 32; ++kb) {
    v16bf a2;
#pragma unroll
    for (int i = 0; i < 16; ++i)
      a2[i] = h_lds[s2 * 16 + m][kb * 32 + akmap(i, hi)];
#pragma unroll
    for (int t = 0; t < 4; ++t) {
      int dt = hd * 4 + t;
      v16bf b = *(const v16bf*)(fwfrag + (((size_t)kb * 8 + dt) * 32 + lane) * 16);
      yacc[t] = __builtin_amdgcn_wmma_f32_16x16x32_bf16(
          false, a2, false, b, (short)0, yacc[t], false, false);
    }
  }
  // bias + relu -> y_lds (bf16)
#pragma unroll
  for (int t = 0; t < 4; ++t) {
    int dt = hd * 4 + t;
    float bias = fb[dt * 16 + m];
#pragma unroll
    for (int g = 0; g < 8; ++g) {
      float v = yacc[t][g] + bias;
      v = v > 0.f ? v : 0.f;
      y_lds[s2 * 16 + g + hi * 8][dt * 16 + m] = (__bf16)v;
    }
  }
  __syncthreads();

  // Cooperative store of y block in B-fragment layout (coalesced 32B/lane).
  for (int gidx = tid; gidx < 512; gidx += 256) {
    int lam = gidx & 31;
    int dt  = (gidx >> 5) & 7;
    int eb2 = gidx >> 8;                       // 0 or 1 (32-e sub-block)
    union { v16bf v; __bf16 s[16]; } tmp;
#pragma unroll
    for (int i = 0; i < 16; ++i)
      tmp.s[i] = y_lds[eb2 * 32 + (i & 7) + ((lam >> 4) << 3) + ((i & 8) << 1)][dt * 16 + (lam & 15)];
    size_t off = (((size_t)(blockIdx.x * 2 + eb2) * 8 + dt) * 32 + lam) * 16;
    *(v16bf*)(yfrag + off) = tmp.v;
  }
}

// ---------------------------------------------------------------------------
// Kernel 3: out_partial[split] += tgt[nblk, esplit] @ y[esplit, :]
// grid (N/128, 8 splits); wave w owns 16 rows x 128 cols; K chunk = 4096.
// ---------------------------------------------------------------------------
__global__ __launch_bounds__(256) void scatter_gemm(
    const float* __restrict__ tgt, const __bf16* __restrict__ yfrag,
    float* __restrict__ partials) {
  const int tid = threadIdx.x, w = tid >> 5, lane = tid & 31;
  const int m = lane & 15, hi = lane >> 4;
  const int n_row = blockIdx.x * 128 + w * 16 + m;
  const int ebase = blockIdx.y * (Ee / 8);

  v8f acc[8] = {};
  for (int eb = 0; eb < (Ee / 8) / 32; ++eb) {
    const int e0 = ebase + eb * 32;
    // A tile (16n x 32e): per-lane 4x b128 along the row (full 128B consumed).
    const float* row = tgt + (size_t)n_row * Ee + e0 + hi * 8;
    f4 c0 = __builtin_nontemporal_load((const f4*)(row));
    f4 c1 = __builtin_nontemporal_load((const f4*)(row + 4));
    f4 c2 = __builtin_nontemporal_load((const f4*)(row + 16));
    f4 c3 = __builtin_nontemporal_load((const f4*)(row + 20));
    v16bf a;
#pragma unroll
    for (int i = 0; i < 4; ++i) {
      a[i]      = (__bf16)c0[i];
      a[i + 4]  = (__bf16)c1[i];
      a[i + 8]  = (__bf16)c2[i];
      a[i + 12] = (__bf16)c3[i];
    }
    const int eg = e0 >> 5;
#pragma unroll
    for (int t = 0; t < 8; ++t) {
      v16bf b = *(const v16bf*)(yfrag + (((size_t)eg * 8 + t) * 32 + lane) * 16);
      acc[t] = __builtin_amdgcn_wmma_f32_16x16x32_bf16(
          false, a, false, b, (short)0, acc[t], false, false);
    }
  }
  float* p = partials + (size_t)blockIdx.y * Nn * DO;
#pragma unroll
  for (int t = 0; t < 8; ++t)
#pragma unroll
    for (int g = 0; g < 8; ++g)
      p[(size_t)(blockIdx.x * 128 + w * 16 + g + hi * 8) * DO + t * 16 + m] = acc[t][g];
}

// ---------------------------------------------------------------------------
// Kernel 4: reduce split-K partials into the f32 output.
// ---------------------------------------------------------------------------
__global__ __launch_bounds__(256) void reduce_partials(
    const float* __restrict__ partials, float* __restrict__ out) {
  int idx = blockIdx.x * 256 + threadIdx.x;
  if (idx < Nn * DO) {
    float s = 0.f;
#pragma unroll
    for (int k = 0; k < 8; ++k) s += partials[(size_t)k * Nn * DO + idx];
    out[idx] = s;
  }
}

extern "C" void kernel_launch(void* const* d_in, const int* in_sizes, int n_in,
                              void* d_out, int out_size, void* d_ws, size_t ws_size,
                              hipStream_t stream) {
  (void)in_sizes; (void)n_in; (void)out_size; (void)ws_size;
  const float* x   = (const float*)d_in[0];
  const float* src = (const float*)d_in[1];
  const float* tgt = (const float*)d_in[2];
  const float* fw  = (const float*)d_in[3];
  const float* fb  = (const float*)d_in[4];
  float* out = (float*)d_out;

  // Workspace layout: [0,1MB) xfrag | [1MB,1.0625MB) fwfrag |
  //                   [2MB,10MB) yfrag bf16 | [16MB,32MB) split-K partials f32
  __bf16* xfrag    = (__bf16*)d_ws;
  __bf16* fwfrag   = (__bf16*)((char*)d_ws + (1u << 20));
  __bf16* yfrag    = (__bf16*)((char*)d_ws + (2u << 20));
  float*  partials = (float*) ((char*)d_ws + (16u << 20));

  prep_frags<<<(524288 + 32768) / 256, 256, 0, stream>>>(x, fw, xfrag, fwfrag);
  gather_mlp<<<Ee / 64, 256, 0, stream>>>(src, tgt, fb, xfrag, fwfrag, yfrag);
  scatter_gemm<<<dim3(Nn / 128, 8), 256, 0, stream>>>(tgt, yfrag, partials);
  reduce_partials<<<(Nn * DO) / 256, 256, 0, stream>>>(partials, out);
}
